// YakMoE_11132555231282
// MI455X (gfx1250) — matrix operand order repaired and checked by reading, not compiled, not run on hardware
//
#include <hip/hip_runtime.h>
#include <hip/hip_bf16.h>

// ---------------------------------------------------------------------------
// MoE (top-2 of 8 experts), T=8192, H=2048, F=4096, fp32 in/out.
// Plan: dense per-expert GEMMs (combine weights are zero for unselected
// experts, so this matches the reference), bf16 WMMA with fp32 accumulation.
// Operands are pre-converted fp32->bf16 ONCE per call (x once; each expert's
// w1/w3/w2 once into rotating ws buffers), so the GEMM K-loop staging is pure
// 16B copies -> no VALU conversion storm; the matrix pipe is the limiter.
// Fallback to on-the-fly conversion if ws_size is too small.
// ---------------------------------------------------------------------------

typedef __attribute__((ext_vector_type(16))) __bf16          v16bf;
typedef __attribute__((ext_vector_type(8)))  __bf16          v8bf;
typedef __attribute__((ext_vector_type(8)))  float           v8f;
typedef __attribute__((ext_vector_type(4)))  float           v4f;
typedef __attribute__((ext_vector_type(8)))  unsigned short  v8u16;

#define LDSK 40   // padded row stride (bf16 elems) for 32-wide K tiles: 80B, conflict-free

static __device__ inline unsigned short f2bfbits(float f) {
    union { __bf16 b; unsigned short u; } c;
    c.b = (__bf16)f;               // native RNE convert (v_cvt_pk_bf16_f32 class)
    return c.u;
}

static __device__ inline v8f vzero() {
    v8f z = {0.f, 0.f, 0.f, 0.f, 0.f, 0.f, 0.f, 0.f};
    return z;
}

// A-matrix 16x32 bf16 fragment (ISA 7.12.2): lane L holds row m=L&15;
// lanes 0-15: K=0..7 (v0-3), K=16..23 (v4-7); lanes 16-31: K=8..15, K=24..31.
static __device__ inline v16bf load_frag_a(const unsigned short* s, int lane) {
    int m = lane & 15, half = lane >> 4;
    const unsigned short* p = s + m * LDSK + half * 8;
    union { v16bf v; v8bf h[2]; } u;
    u.h[0] = *(const v8bf*)(p);
    u.h[1] = *(const v8bf*)(p + 16);
    return u.v;
}

// B-matrix 32x16 bf16 fragment, B[k][n] = W[n][k] (row-major W tile in LDS):
// lane L holds col n=L&15; lanes 0-15: K=0..7, K=16..23; lanes 16-31: K=8..15, K=24..31.
static __device__ inline v16bf load_frag_b(const unsigned short* s, int lane) {
    int n = lane & 15, half = lane >> 4;
    const unsigned short* p = s + n * LDSK + half * 8;
    union { v16bf v; v8bf h[2]; } u;
    u.h[0] = *(const v8bf*)(p);
    u.h[1] = *(const v8bf*)(p + 16);
    return u.v;
}

static __device__ inline v8f wmma_bf16(v16bf a, v16bf b, v8f c) {
    return __builtin_amdgcn_wmma_f32_16x16x32_bf16(
        false, a, false, b, (short)0, c, false, false);
}

// ---------------------------------------------------------------------------
// fp32 -> bf16 bulk converter: 8 elements/thread, streaming (non-temporal)
// reads so the one-shot fp32 traffic does not evict the bf16 L2 working set.
// n8 = element_count / 8 (all sizes here are multiples of 8).
// ---------------------------------------------------------------------------
__global__ __launch_bounds__(256) void cvt_f32_bf16(const float* __restrict__ in,
                                                    unsigned short* __restrict__ out,
                                                    long long n8) {
    long long i = (long long)blockIdx.x * 256 + threadIdx.x;
    if (i >= n8) return;
    const v4f* p = (const v4f*)in + i * 2;
    v4f a = __builtin_nontemporal_load(p);
    v4f b = __builtin_nontemporal_load(p + 1);
    v8u16 o;
    o[0] = f2bfbits(a[0]); o[1] = f2bfbits(a[1]); o[2] = f2bfbits(a[2]); o[3] = f2bfbits(a[3]);
    o[4] = f2bfbits(b[0]); o[5] = f2bfbits(b[1]); o[6] = f2bfbits(b[2]); o[7] = f2bfbits(b[3]);
    *(v8u16*)(out + i * 8) = o;
}

// ---------------------------------------------------------------------------
// Router: logits = x @ gate_w^T, softmax(fp32), top-2, renormalize -> dense
// combine[T][8]. One 256-thread block per token.
// ---------------------------------------------------------------------------
__global__ __launch_bounds__(256) void moe_router(const float* __restrict__ x,
                                                  const float* __restrict__ gw,
                                                  float* __restrict__ combine,
                                                  int H) {
    __shared__ float red[8 * 256];
    const int t = blockIdx.x, tid = threadIdx.x;
    float p[8];
#pragma unroll
    for (int e = 0; e < 8; ++e) p[e] = 0.f;
    const float* xr = x + (size_t)t * H;
    for (int i = tid; i < H; i += 256) {
        float xv = xr[i];
#pragma unroll
        for (int e = 0; e < 8; ++e) p[e] += xv * gw[(size_t)e * H + i];
    }
#pragma unroll
    for (int e = 0; e < 8; ++e) red[e * 256 + tid] = p[e];
    __syncthreads();
    for (int sft = 128; sft > 0; sft >>= 1) {
        if (tid < sft) {
#pragma unroll
            for (int e = 0; e < 8; ++e) red[e * 256 + tid] += red[e * 256 + tid + sft];
        }
        __syncthreads();
    }
    if (tid == 0) {
        float l[8], mx = red[0];
#pragma unroll
        for (int e = 0; e < 8; ++e) { l[e] = red[e * 256]; mx = fmaxf(mx, l[e]); }
        float sum = 0.f;
#pragma unroll
        for (int e = 0; e < 8; ++e) { l[e] = __expf(l[e] - mx); sum += l[e]; }
        float inv = 1.f / sum;
#pragma unroll
        for (int e = 0; e < 8; ++e) l[e] *= inv;
        int i0 = 0;
#pragma unroll
        for (int e = 1; e < 8; ++e) if (l[e] > l[i0]) i0 = e;
        int i1 = (i0 == 0) ? 1 : 0;
#pragma unroll
        for (int e = 0; e < 8; ++e) if (e != i0 && l[e] > l[i1]) i1 = e;
        float norm = 1.f / (l[i0] + l[i1]);
#pragma unroll
        for (int e = 0; e < 8; ++e) {
            float c = (e == i0) ? l[i0] * norm : ((e == i1) ? l[i1] * norm : 0.f);
            combine[(size_t)t * 8 + e] = c;
        }
    }
}

// ---------------------------------------------------------------------------
// Gate+Up fused (bf16 operands): h = silu(xb @ w1b^T) * (xb @ w3b^T).
// Block tile 128(T) x 64(F), K over H in steps of 32. 8 waves = 4(M) x 2(N);
// wave tile 32x32 -> 2 A-frags shared by 4 B-frags = 8 WMMA / wave / K-step.
// ---------------------------------------------------------------------------
__global__ __launch_bounds__(256) void moe_gateup_bf16(
        const unsigned short* __restrict__ xb,
        const unsigned short* __restrict__ w1b,
        const unsigned short* __restrict__ w3b,
        unsigned short* __restrict__ hbuf, int H, int F) {
    __shared__ unsigned short sA[128 * LDSK];
    __shared__ unsigned short sB1[64 * LDSK];
    __shared__ unsigned short sB3[64 * LDSK];
    const int tid = threadIdx.x;
    const int lane = tid & 31, wave = tid >> 5;
    const int m0 = (wave & 3) * 32, n0 = (wave >> 2) * 32;
    const int rowBase = blockIdx.y * 128;     // token rows
    const int colBase = blockIdx.x * 64;      // F columns

    v8f acc1[2][2], acc3[2][2];
#pragma unroll
    for (int mi = 0; mi < 2; ++mi)
#pragma unroll
        for (int ni = 0; ni < 2; ++ni) { acc1[mi][ni] = vzero(); acc3[mi][ni] = vzero(); }

    for (int kb = 0; kb < H; kb += 32) {
        {   // A tile 128x32: 2 threads/row, 16 elems (32B) each
            int r = tid >> 1, c = (tid & 1) << 4;
            const v8u16* g = (const v8u16*)(xb + (size_t)(rowBase + r) * H + kb + c);
            v8u16* d = (v8u16*)(sA + r * LDSK + c);
            d[0] = g[0]; d[1] = g[1];
        }
        {   // B tiles 64x32: 4 threads/row, 8 elems (16B) each
            int r = tid >> 2, c = (tid & 3) << 3;
            *(v8u16*)(sB1 + r * LDSK + c) =
                *(const v8u16*)(w1b + (size_t)(colBase + r) * H + kb + c);
            *(v8u16*)(sB3 + r * LDSK + c) =
                *(const v8u16*)(w3b + (size_t)(colBase + r) * H + kb + c);
        }
        __syncthreads();
        v16bf a0  = load_frag_a(sA + m0 * LDSK, lane);
        v16bf a1  = load_frag_a(sA + (m0 + 16) * LDSK, lane);
        v16bf b10 = load_frag_b(sB1 + n0 * LDSK, lane);
        v16bf b11 = load_frag_b(sB1 + (n0 + 16) * LDSK, lane);
        v16bf b30 = load_frag_b(sB3 + n0 * LDSK, lane);
        v16bf b31 = load_frag_b(sB3 + (n0 + 16) * LDSK, lane);
        acc1[0][0] = wmma_bf16(a0, b10, acc1[0][0]);
        acc1[0][1] = wmma_bf16(a0, b11, acc1[0][1]);
        acc1[1][0] = wmma_bf16(a1, b10, acc1[1][0]);
        acc1[1][1] = wmma_bf16(a1, b11, acc1[1][1]);
        acc3[0][0] = wmma_bf16(a0, b30, acc3[0][0]);
        acc3[0][1] = wmma_bf16(a0, b31, acc3[0][1]);
        acc3[1][0] = wmma_bf16(a1, b30, acc3[1][0]);
        acc3[1][1] = wmma_bf16(a1, b31, acc3[1][1]);
        __syncthreads();
    }

    const int half8 = (lane >> 4) * 8, ncol = lane & 15;
#pragma unroll
    for (int mi = 0; mi < 2; ++mi)
#pragma unroll
        for (int ni = 0; ni < 2; ++ni)
#pragma unroll
            for (int r = 0; r < 8; ++r) {
                float g = acc1[mi][ni][r];
                float u = acc3[mi][ni][r];
                float hv = (g / (1.f + __expf(-g))) * u;   // silu(g) * u
                int m = m0 + mi * 16 + half8 + r;
                int n = n0 + ni * 16 + ncol;
                hbuf[(size_t)(rowBase + m) * F + colBase + n] = f2bfbits(hv);
            }
}

// ---------------------------------------------------------------------------
// Down-proj + combine (bf16 operands): out (+)= combine[:,e] * (hb @ w2b^T).
// Block tile 128(T) x 64(H), K over F in steps of 32; wave tile 32x32.
// ---------------------------------------------------------------------------
__global__ __launch_bounds__(256) void moe_down_bf16(
        const unsigned short* __restrict__ hb,
        const unsigned short* __restrict__ w2b,
        const float* __restrict__ combine, float* __restrict__ out,
        int H, int F, int e, int accumulate) {
    __shared__ unsigned short sA[128 * LDSK];
    __shared__ unsigned short sB[64 * LDSK];
    __shared__ float sC[128];
    const int tid = threadIdx.x;
    const int lane = tid & 31, wave = tid >> 5;
    const int m0 = (wave & 3) * 32, n0 = (wave >> 2) * 32;
    const int rowBase = blockIdx.y * 128;
    const int colBase = blockIdx.x * 64;

    if (tid < 128) sC[tid] = combine[(size_t)(rowBase + tid) * 8 + e];

    v8f acc[2][2];
#pragma unroll
    for (int mi = 0; mi < 2; ++mi)
#pragma unroll
        for (int ni = 0; ni < 2; ++ni) acc[mi][ni] = vzero();

    for (int kb = 0; kb < F; kb += 32) {
        {   // A tile 128x32 bf16 copy
            int r = tid >> 1, c = (tid & 1) << 4;
            const v8u16* g = (const v8u16*)(hb + (size_t)(rowBase + r) * F + kb + c);
            v8u16* d = (v8u16*)(sA + r * LDSK + c);
            d[0] = g[0]; d[1] = g[1];
        }
        {   // B tile 64x32 bf16 copy
            int r = tid >> 2, c = (tid & 3) << 3;
            *(v8u16*)(sB + r * LDSK + c) =
                *(const v8u16*)(w2b + (size_t)(colBase + r) * F + kb + c);
        }
        __syncthreads();
        v16bf a0 = load_frag_a(sA + m0 * LDSK, lane);
        v16bf a1 = load_frag_a(sA + (m0 + 16) * LDSK, lane);
        v16bf b0 = load_frag_b(sB + n0 * LDSK, lane);
        v16bf b1 = load_frag_b(sB + (n0 + 16) * LDSK, lane);
        acc[0][0] = wmma_bf16(a0, b0, acc[0][0]);
        acc[0][1] = wmma_bf16(a0, b1, acc[0][1]);
        acc[1][0] = wmma_bf16(a1, b0, acc[1][0]);
        acc[1][1] = wmma_bf16(a1, b1, acc[1][1]);
        __syncthreads();
    }

    const int half8 = (lane >> 4) * 8, ncol = lane & 15;
#pragma unroll
    for (int mi = 0; mi < 2; ++mi)
#pragma unroll
        for (int ni = 0; ni < 2; ++ni)
#pragma unroll
            for (int r = 0; r < 8; ++r) {
                int m = m0 + mi * 16 + half8 + r;
                float v = sC[m] * acc[mi][ni][r];
                size_t oi = (size_t)(rowBase + m) * H + colBase + n0 + ni * 16 + ncol;
                if (accumulate) out[oi] += v;
                else            out[oi] = v;
            }
}

// ---------------------------------------------------------------------------
// Fallback kernels (on-the-fly fp32->bf16 in LDS) for small workspaces.
// ---------------------------------------------------------------------------
static __device__ inline void stage_f32_bf16(const float* __restrict__ g, size_t ldg,
                                             unsigned short* __restrict__ s, int tid) {
    int r = tid >> 2;
    int c = (tid & 3) << 3;
    const v4f* p = (const v4f*)(g + (size_t)r * ldg + c);
    v4f f0 = p[0], f1 = p[1];
    v8u16 o;
    o[0] = f2bfbits(f0[0]); o[1] = f2bfbits(f0[1]); o[2] = f2bfbits(f0[2]); o[3] = f2bfbits(f0[3]);
    o[4] = f2bfbits(f1[0]); o[5] = f2bfbits(f1[1]); o[6] = f2bfbits(f1[2]); o[7] = f2bfbits(f1[3]);
    *(v8u16*)(s + r * LDSK + c) = o;
}

__global__ __launch_bounds__(256) void moe_gateup_f32(const float* __restrict__ x,
                                                      const float* __restrict__ w1e,
                                                      const float* __restrict__ w3e,
                                                      unsigned short* __restrict__ hbuf,
                                                      int H, int F) {
    __shared__ unsigned short sA[64 * LDSK];
    __shared__ unsigned short sB1[64 * LDSK];
    __shared__ unsigned short sB3[64 * LDSK];
    const int tid = threadIdx.x;
    const int lane = tid & 31, wave = tid >> 5;
    const int m0 = (wave & 3) * 16, n0 = (wave >> 2) * 32;
    const int rowBase = blockIdx.y * 64, colBase = blockIdx.x * 64;
    v8f a1[2] = {vzero(), vzero()}, a3[2] = {vzero(), vzero()};
    for (int kb = 0; kb < H; kb += 32) {
        stage_f32_bf16(x   + (size_t)rowBase * H + kb, (size_t)H, sA,  tid);
        stage_f32_bf16(w1e + (size_t)colBase * H + kb, (size_t)H, sB1, tid);
        stage_f32_bf16(w3e + (size_t)colBase * H + kb, (size_t)H, sB3, tid);
        __syncthreads();
        v16bf a   = load_frag_a(sA  + m0 * LDSK, lane);
        v16bf b10 = load_frag_b(sB1 + n0 * LDSK, lane);
        v16bf b11 = load_frag_b(sB1 + (n0 + 16) * LDSK, lane);
        v16bf b30 = load_frag_b(sB3 + n0 * LDSK, lane);
        v16bf b31 = load_frag_b(sB3 + (n0 + 16) * LDSK, lane);
        a1[0] = wmma_bf16(a, b10, a1[0]);
        a1[1] = wmma_bf16(a, b11, a1[1]);
        a3[0] = wmma_bf16(a, b30, a3[0]);
        a3[1] = wmma_bf16(a, b31, a3[1]);
        __syncthreads();
    }
    const int half8 = (lane >> 4) * 8, ncol = lane & 15;
#pragma unroll
    for (int j = 0; j < 2; ++j)
#pragma unroll
        for (int r = 0; r < 8; ++r) {
            float g = a1[j][r], u = a3[j][r];
            float hv = (g / (1.f + __expf(-g))) * u;
            hbuf[(size_t)(rowBase + m0 + half8 + r) * F + colBase + n0 + j * 16 + ncol] =
                f2bfbits(hv);
        }
}

__global__ __launch_bounds__(256) void moe_down_f32(const unsigned short* __restrict__ hbuf,
                                                    const float* __restrict__ w2e,
                                                    const float* __restrict__ combine,
                                                    float* __restrict__ out,
                                                    int H, int F, int e, int accumulate) {
    __shared__ unsigned short sA[64 * LDSK];
    __shared__ unsigned short sB[64 * LDSK];
    __shared__ float sC[64];
    const int tid = threadIdx.x;
    const int lane = tid & 31, wave = tid >> 5;
    const int m0 = (wave & 3) * 16, n0 = (wave >> 2) * 32;
    const int rowBase = blockIdx.y * 64, colBase = blockIdx.x * 64;
    if (tid < 64) sC[tid] = combine[(size_t)(rowBase + tid) * 8 + e];
    v8f acc[2] = {vzero(), vzero()};
    for (int kb = 0; kb < F; kb += 32) {
        {
            int r = tid >> 2, c = (tid & 3) << 3;
            *(v8u16*)(sA + r * LDSK + c) =
                *(const v8u16*)(hbuf + (size_t)(rowBase + r) * F + kb + c);
        }
        stage_f32_bf16(w2e + (size_t)colBase * F + kb, (size_t)F, sB, tid);
        __syncthreads();
        v16bf a  = load_frag_a(sA + m0 * LDSK, lane);
        v16bf b0 = load_frag_b(sB + n0 * LDSK, lane);
        v16bf b1 = load_frag_b(sB + (n0 + 16) * LDSK, lane);
        acc[0] = wmma_bf16(a, b0, acc[0]);
        acc[1] = wmma_bf16(a, b1, acc[1]);
        __syncthreads();
    }
    const int half8 = (lane >> 4) * 8, ncol = lane & 15;
#pragma unroll
    for (int j = 0; j < 2; ++j)
#pragma unroll
        for (int r = 0; r < 8; ++r) {
            int m = m0 + half8 + r;
            float v = sC[m] * acc[j][r];
            size_t oi = (size_t)(rowBase + m) * H + colBase + n0 + j * 16 + ncol;
            if (accumulate) out[oi] += v;
            else            out[oi] = v;
        }
}

// ---------------------------------------------------------------------------
extern "C" void kernel_launch(void* const* d_in, const int* in_sizes, int n_in,
                              void* d_out, int out_size, void* d_ws, size_t ws_size,
                              hipStream_t stream) {
    const float* x  = (const float*)d_in[0];   // [B,S,H] fp32
    const float* gw = (const float*)d_in[1];   // [E,H]
    const float* w1 = (const float*)d_in[2];   // [E,F,H]
    const float* w2 = (const float*)d_in[3];   // [E,H,F]
    const float* w3 = (const float*)d_in[4];   // [E,F,H]
    float* out = (float*)d_out;

    const int E = 8;
    const int H = in_sizes[1] / E;              // 2048
    const int T = in_sizes[0] / H;              // 8192
    const int F = in_sizes[2] / (E * H);        // 4096
    const size_t FH = (size_t)F * H;

    auto al = [](size_t b) { return (b + 255) & ~(size_t)255; };
    size_t comb_b = al((size_t)T * E * sizeof(float));
    size_t xb_b   = al((size_t)T * H * 2);
    size_t hb_b   = al((size_t)T * F * 2);
    size_t wsl_b  = al(FH * 2);                 // one bf16 weight slice

    float* combine = (float*)d_ws;

    if (ws_size >= comb_b + xb_b + hb_b + 3 * wsl_b) {
        // Fast path: pre-converted bf16 operands.
        char* base = (char*)d_ws;
        unsigned short* xb  = (unsigned short*)(base + comb_b);
        unsigned short* hb  = (unsigned short*)(base + comb_b + xb_b);
        unsigned short* wb1 = (unsigned short*)(base + comb_b + xb_b + hb_b);
        unsigned short* wb3 = (unsigned short*)(base + comb_b + xb_b + hb_b + wsl_b);
        unsigned short* wb2 = (unsigned short*)(base + comb_b + xb_b + hb_b + 2 * wsl_b);

        moe_router<<<dim3(T), dim3(256), 0, stream>>>(x, gw, combine, H);

        long long xn8 = (long long)T * H / 8;
        cvt_f32_bf16<<<dim3((unsigned)((xn8 + 255) / 256)), dim3(256), 0, stream>>>(x, xb, xn8);

        long long wn8 = (long long)FH / 8;
        dim3 cvtGrid((unsigned)((wn8 + 255) / 256));
        dim3 gridGU(F / 64, T / 128);
        dim3 gridDN(H / 64, T / 128);
        for (int e = 0; e < E; ++e) {
            cvt_f32_bf16<<<cvtGrid, dim3(256), 0, stream>>>(w1 + (size_t)e * FH, wb1, wn8);
            cvt_f32_bf16<<<cvtGrid, dim3(256), 0, stream>>>(w3 + (size_t)e * FH, wb3, wn8);
            cvt_f32_bf16<<<cvtGrid, dim3(256), 0, stream>>>(w2 + (size_t)e * FH, wb2, wn8);
            moe_gateup_bf16<<<gridGU, dim3(256), 0, stream>>>(xb, wb1, wb3, hb, H, F);
            moe_down_bf16<<<gridDN, dim3(256), 0, stream>>>(hb, wb2, combine, out,
                                                            H, F, e, e > 0 ? 1 : 0);
        }
    } else {
        // Fallback: on-the-fly conversion, ws = combine + h only.
        unsigned short* hb = (unsigned short*)((char*)d_ws + comb_b);
        moe_router<<<dim3(T), dim3(256), 0, stream>>>(x, gw, combine, H);
        dim3 gridGU(F / 64, T / 64);
        dim3 gridDN(H / 64, T / 64);
        for (int e = 0; e < E; ++e) {
            moe_gateup_f32<<<gridGU, dim3(256), 0, stream>>>(x, w1 + (size_t)e * FH,
                                                             w3 + (size_t)e * FH, hb, H, F);
            moe_down_f32<<<gridDN, dim3(256), 0, stream>>>(hb, w2 + (size_t)e * FH,
                                                           combine, out, H, F, e,
                                                           e > 0 ? 1 : 0);
        }
    }
}
